// HierarchicalBatchTopKSAE_singleTopK_84146999263584
// MI455X (gfx1250) — matrix operand order repaired and checked by reading, not compiled, not run on hardware
//
#include <hip/hip_runtime.h>
#include <stdint.h>

// ---------------- problem constants ----------------
#define BATCH   4096
#define ADIM    768
#define S0N     4096
#define DICT    32768
#define KTOT    (64 * BATCH)     // global top-k count = 262144
#define CAP     KTOT
#define HBINS   4096

typedef __bf16 bf16;
typedef bf16  v16bf __attribute__((ext_vector_type(16)));
typedef float v8f   __attribute__((ext_vector_type(8)));

union Frag16 { v16bf v; uint4 u[2]; };

// ---------------- workspace layout (bytes) ----------------
static constexpr size_t OFF_XC    = 0;
static constexpr size_t OFF_W0B   = OFF_XC    + (size_t)BATCH * ADIM * 2;
static constexpr size_t OFF_W1B   = OFF_W0B   + (size_t)S0N   * ADIM * 2;
static constexpr size_t OFF_WDTB  = OFF_W1B   + (size_t)DICT  * ADIM * 2;
static constexpr size_t OFF_L0    = OFF_WDTB  + (size_t)DICT  * ADIM * 2;
static constexpr size_t OFF_FINAL = OFF_L0    + (size_t)BATCH * S0N  * 4;
static constexpr size_t OFF_HIST  = OFF_FINAL + (size_t)BATCH * DICT * 2;
static constexpr size_t OFF_RCNT  = OFF_HIST  + (size_t)HBINS * 4;
static constexpr size_t OFF_RSTART= OFF_RCNT  + (size_t)BATCH * 4;
static constexpr size_t OFF_RFILL = OFF_RSTART+ (size_t)(BATCH + 64) * 4;
static constexpr size_t OFF_CNT   = OFF_RFILL + (size_t)BATCH * 4;
static constexpr size_t OFF_LIDX  = OFF_CNT   + 256;
static constexpr size_t OFF_LVAL  = OFF_LIDX  + (size_t)CAP * 4;
static constexpr size_t OFF_CCOL  = OFF_LVAL  + (size_t)CAP * 4;
static constexpr size_t OFF_CVAL  = OFF_CCOL  + (size_t)CAP * 4;

// ---------------- CDNA5 async helpers ----------------
__device__ __forceinline__ void async_copy_b128(const void* gsrc, void* lds_dst) {
  // Per-lane Global->LDS DMA, tracked by ASYNCcnt (no VGPR staging).
  const unsigned lds_off = (unsigned)(uintptr_t)lds_dst;
  asm volatile("global_load_async_to_lds_b128 %0, %1, off"
               :: "v"(lds_off), "v"(gsrc) : "memory");
}

__device__ __forceinline__ void wait_async0() {
#if __has_builtin(__builtin_amdgcn_s_wait_asynccnt)
  __builtin_amdgcn_s_wait_asynccnt(0);
#else
  asm volatile("s_wait_asynccnt 0x0" ::: "memory");
#endif
}

// ---------------- prep kernels ----------------
__global__ void prep_xc(const float* __restrict__ x, const float* __restrict__ bdec,
                        bf16* __restrict__ xc) {
  const long n = (long)BATCH * ADIM;
  for (long i = (long)blockIdx.x * blockDim.x + threadIdx.x; i < n;
       i += (long)gridDim.x * blockDim.x)
    xc[i] = (bf16)(x[i] - bdec[i % ADIM]);
}

__global__ void conv_bf16(const float* __restrict__ src, bf16* __restrict__ dst, long n) {
  for (long i = (long)blockIdx.x * blockDim.x + threadIdx.x; i < n;
       i += (long)gridDim.x * blockDim.x)
    dst[i] = (bf16)src[i];
}

// Wd [ADIM, DICT] f32 -> WdT [DICT, ADIM] bf16, LDS-tiled so both sides coalesce
__global__ void transpose_wd(const float* __restrict__ Wd, bf16* __restrict__ WdT) {
  __shared__ float tile[32][33];
  const int tx = threadIdx.x, ty = threadIdx.y;
  const int j0 = blockIdx.x * 32, d0 = blockIdx.y * 32;
  for (int i = ty; i < 32; i += 8)
    tile[i][tx] = Wd[(size_t)(d0 + i) * DICT + j0 + tx];
  __syncthreads();
  for (int i = ty; i < 32; i += 8)
    WdT[(size_t)(j0 + i) * ADIM + d0 + tx] = (bf16)tile[tx][i];
}

__global__ void init_zero(unsigned* __restrict__ hist, unsigned* __restrict__ rowCnt,
                          unsigned* __restrict__ rowFill, unsigned* __restrict__ cnt) {
  const int i = blockIdx.x * blockDim.x + threadIdx.x;
  if (i < HBINS) hist[i] = 0;
  if (i < BATCH) { rowCnt[i] = 0; rowFill[i] = 0; }
  if (i < 64) cnt[i] = 0;
}

// ---------------- main encoder GEMM (WMMA bf16, async double-buffered) ----------------
// 128-thread blocks = 4 waves in a 2x2 grid; each wave owns a 64x64 output
// tile (4x4 WMMA accumulators) -> 16 ds_load_b128 feed 16 WMMAs per K-step.
#define BM 128
#define BN 128
#define BK 32
#define NKT (ADIM / BK)   // 24 K-tiles
#define LDT 40            // LDS row stride in bf16 (80 B, 16B-aligned, bank-skewed)
#define GTHREADS 128

template<bool GATE>
__global__ __launch_bounds__(GTHREADS)
void sae_encoder_gemm(const bf16* __restrict__ A, const bf16* __restrict__ W,
                      const float* __restrict__ bias, const float* __restrict__ l0,
                      float* __restrict__ outF, bf16* __restrict__ outB,
                      unsigned* __restrict__ gHist, int N)
{
  __shared__ bf16 As[2][BM * LDT];
  __shared__ bf16 Bs[2][BN * LDT];
  __shared__ unsigned histS[GATE ? HBINS : 1];

  const int t    = threadIdx.x;
  const int wave = t >> 5;
  const int lane = t & 31;
  const int waveM = wave >> 1;   // 0..1 -> 64-row slab
  const int waveN = wave & 1;    // 0..1 -> 64-col slab
  const int lm  = lane & 15;
  const int lhi = lane >> 4;     // lane half selects K group per ISA layout

  const int blockN = blockIdx.x * BN;
  const int blockM = blockIdx.y * BM;

  if constexpr (GATE) {
    for (int i = t; i < HBINS; i += GTHREADS) histS[i] = 0;
  }

  // per-thread tile-chunk coordinates: 512 16B chunks per tile, 4 per thread
  const int q = t & 3;                      // 16B chunk within a 64B row slice
  int crow[4];
  const bf16* gA[4];
  const bf16* gW[4];
  unsigned loffA[4], loffB[4];
  #pragma unroll
  for (int s = 0; s < 4; ++s) {
    const int c = t + s * GTHREADS;         // 0..511
    crow[s] = c >> 2;
    gA[s] = A + (size_t)(blockM + crow[s]) * ADIM + q * 8;
    gW[s] = W + (size_t)(blockN + crow[s]) * ADIM + q * 8;
    loffA[s] = (unsigned)(crow[s] * LDT + q * 8);
    loffB[s] = loffA[s];
  }

  v8f acc[4][4];
  #pragma unroll
  for (int i = 0; i < 4; ++i)
    #pragma unroll
    for (int j = 0; j < 4; ++j)
      #pragma unroll
      for (int e = 0; e < 8; ++e) acc[i][j][e] = 0.f;

  // prologue: async-fill buffer 0 with K-tile 0
  #pragma unroll
  for (int s = 0; s < 4; ++s) {
    async_copy_b128(gA[s], &As[0][loffA[s]]);
    async_copy_b128(gW[s], &Bs[0][loffB[s]]);
  }

  for (int kt = 0; kt < NKT; ++kt) {
    const int cur = kt & 1;
    wait_async0();        // my copies into `cur` are in LDS
    __syncthreads();      // everyone's copies landed; prior reads of `nxt` done

    if (kt + 1 < NKT) {   // DMA next K-tile into the other buffer
      const int nxt = cur ^ 1;
      const int ko = (kt + 1) * BK;
      #pragma unroll
      for (int s = 0; s < 4; ++s) {
        async_copy_b128(gA[s] + ko, &As[nxt][loffA[s]]);
        async_copy_b128(gW[s] + ko, &Bs[nxt][loffB[s]]);
      }
    }

    // A frag (16x32 bf16): lanes<16 take K 0-7 & 16-23, lanes>=16 K 8-15 & 24-31
    Frag16 aF[4], bF;
    #pragma unroll
    for (int tm = 0; tm < 4; ++tm) {
      const bf16* p = &As[cur][(waveM * 64 + tm * 16 + lm) * LDT + lhi * 8];
      aF[tm].u[0] = *(const uint4*)p;
      aF[tm].u[1] = *(const uint4*)(p + 16);
    }
    // B frag (32x16 bf16): lanes<16 take K 0-7 & 16-23, lanes>=16 K 8-15 & 24-31
    #pragma unroll
    for (int tn = 0; tn < 4; ++tn) {
      const bf16* p = &Bs[cur][(waveN * 64 + tn * 16 + lm) * LDT + lhi * 8];
      bF.u[0] = *(const uint4*)p;
      bF.u[1] = *(const uint4*)(p + 16);
      #pragma unroll
      for (int tm = 0; tm < 4; ++tm)
        acc[tm][tn] = __builtin_amdgcn_wmma_f32_16x16x32_bf16(
            false, aF[tm].v, false, bF.v, (short)0, acc[tm][tn], false, false);
    }
  }

  // epilogue: C layout = lane(N 0-15), vgpr r -> M = r + 8*laneHalf
  #pragma unroll
  for (int tm = 0; tm < 4; ++tm) {
    #pragma unroll
    for (int tn = 0; tn < 4; ++tn) {
      const int n_g = blockN + waveN * 64 + tn * 16 + lm;
      const float bv = bias[n_g];
      #pragma unroll
      for (int r = 0; r < 8; ++r) {
        const int m_g = blockM + waveM * 64 + tm * 16 + lhi * 8 + r;
        float v = acc[tm][tn][r] + bv;
        v = v > 0.f ? v : 0.f;
        if constexpr (!GATE) {
          outF[(size_t)m_g * N + n_g] = v;
        } else {
          const float g = v * l0[(size_t)m_g * S0N + (n_g >> 3)];
          const bf16 h = (bf16)g;
          const unsigned short bits = __builtin_bit_cast(unsigned short, h);
          outB[(size_t)m_g * DICT + n_g] = h;
          if (bits != 0 && bits < 0x8000u)      // positive, nonzero
            atomicAdd(&histS[bits >> 4], 1u);
        }
      }
    }
  }
  if constexpr (GATE) {
    __syncthreads();
    for (int i = t; i < HBINS; i += GTHREADS) {
      const unsigned c = histS[i];
      if (c) atomicAdd(&gHist[i], c);
    }
  }
}

// ---------------- threshold / select / CSR / decode ----------------
__global__ void scan_hist(const unsigned* __restrict__ hist, unsigned* __restrict__ cnt) {
  if (threadIdx.x == 0 && blockIdx.x == 0) {
    unsigned cum = 0, thr = 0, remain = KTOT;
    for (int bin = HBINS - 1; bin >= 0; --bin) {
      const unsigned c = hist[bin];
      if (cum + c >= KTOT) { thr = (unsigned)bin; remain = KTOT - cum; break; }
      cum += c;
    }
    cnt[2] = thr; cnt[3] = remain;
  }
}

__global__ void select_topk(const uint4* __restrict__ finalU4, unsigned* __restrict__ cnt,
                            unsigned* __restrict__ rowCnt, unsigned* __restrict__ listIdx,
                            float* __restrict__ listVal) {
  const unsigned thr = cnt[2];
  const unsigned remain = cnt[3];
  const long total = (long)BATCH * DICT / 8;           // 16.7M uint4 chunks
  for (long c = (long)blockIdx.x * blockDim.x + threadIdx.x; c < total;
       c += (long)gridDim.x * blockDim.x) {
    const uint4 u = finalU4[c];
    const unsigned ww[4] = {u.x, u.y, u.z, u.w};
    #pragma unroll
    for (int w = 0; w < 4; ++w) {
      #pragma unroll
      for (int h = 0; h < 2; ++h) {
        const unsigned bits = (ww[w] >> (16 * h)) & 0xFFFFu;
        if (bits == 0 || bits >= 0x8000u) continue;
        const unsigned bin = bits >> 4;
        bool sel = bin > thr;
        if (!sel && bin == thr)
          sel = atomicAdd(&cnt[1], 1u) < remain;       // cap threshold-bin ties at k
        if (sel) {
          const unsigned pos = atomicAdd(&cnt[0], 1u);
          if (pos < CAP) {
            const long f = c * 8 + w * 2 + h;
            const unsigned b = (unsigned)(f >> 15);
            const unsigned j = (unsigned)(f & 32767);
            listIdx[pos] = (b << 15) | j;
            listVal[pos] = __builtin_bit_cast(float, bits << 16);
            atomicAdd(&rowCnt[b], 1u);
          }
        }
      }
    }
  }
}

__global__ void scan_rows(const unsigned* __restrict__ rowCnt, unsigned* __restrict__ rowStart) {
  if (threadIdx.x == 0 && blockIdx.x == 0) {
    unsigned run = 0;
    for (int r = 0; r < BATCH; ++r) { rowStart[r] = run; run += rowCnt[r]; }
    rowStart[BATCH] = run;
  }
}

__global__ void scatter_csr(const unsigned* __restrict__ cnt, const unsigned* __restrict__ listIdx,
                            const float* __restrict__ listVal, const unsigned* __restrict__ rowStart,
                            unsigned* __restrict__ rowFill, unsigned* __restrict__ csrCol,
                            float* __restrict__ csrVal) {
  unsigned total = cnt[0]; if (total > CAP) total = CAP;
  for (unsigned i = blockIdx.x * blockDim.x + threadIdx.x; i < total;
       i += gridDim.x * blockDim.x) {
    const unsigned pk = listIdx[i];
    const unsigned b = pk >> 15, j = pk & 32767u;
    const unsigned slot = rowStart[b] + atomicAdd(&rowFill[b], 1u);
    csrCol[slot] = j;
    csrVal[slot] = listVal[i];
  }
}

// x_hat[b,:] = sum_j val * WdT[j,:] + b_dec  (one block per row; ~64 nnz/row)
__global__ __launch_bounds__(256)
void decode_rows(const unsigned* __restrict__ rowStart, const unsigned* __restrict__ rowCnt,
                 const unsigned* __restrict__ csrCol, const float* __restrict__ csrVal,
                 const bf16* __restrict__ WdT, const float* __restrict__ bdec,
                 float* __restrict__ out) {
  const int b = blockIdx.x;
  const int t = threadIdx.x;
  const unsigned s = rowStart[b];
  const unsigned n = rowCnt[b];
  float a0 = 0.f, a1 = 0.f, a2 = 0.f;
  for (unsigned e = 0; e < n; ++e) {
    const unsigned j = csrCol[s + e];
    const float v = csrVal[s + e];
    const bf16* wr = WdT + (size_t)j * ADIM;
    a0 += v * (float)wr[t];
    a1 += v * (float)wr[t + 256];
    a2 += v * (float)wr[t + 512];
  }
  float* o = out + (size_t)b * ADIM;
  o[t]       = a0 + bdec[t];
  o[t + 256] = a1 + bdec[t + 256];
  o[t + 512] = a2 + bdec[t + 512];
}

// ---------------- host launch ----------------
extern "C" void kernel_launch(void* const* d_in, const int* in_sizes, int n_in,
                              void* d_out, int out_size, void* d_ws, size_t ws_size,
                              hipStream_t stream) {
  (void)in_sizes; (void)n_in; (void)out_size; (void)ws_size;
  const float* x    = (const float*)d_in[0];
  const float* W0   = (const float*)d_in[1];
  const float* b0   = (const float*)d_in[2];
  const float* W1   = (const float*)d_in[3];
  const float* b1   = (const float*)d_in[4];
  const float* Wd   = (const float*)d_in[5];
  const float* bdec = (const float*)d_in[6];
  float* out = (float*)d_out;

  char* ws = (char*)d_ws;
  bf16*     xc     = (bf16*)(ws + OFF_XC);
  bf16*     w0b    = (bf16*)(ws + OFF_W0B);
  bf16*     w1b    = (bf16*)(ws + OFF_W1B);
  bf16*     wdtb   = (bf16*)(ws + OFF_WDTB);
  float*    l0buf  = (float*)(ws + OFF_L0);
  bf16*     finalB = (bf16*)(ws + OFF_FINAL);
  unsigned* hist   = (unsigned*)(ws + OFF_HIST);
  unsigned* rowCnt = (unsigned*)(ws + OFF_RCNT);
  unsigned* rowSt  = (unsigned*)(ws + OFF_RSTART);
  unsigned* rowFl  = (unsigned*)(ws + OFF_RFILL);
  unsigned* cnt    = (unsigned*)(ws + OFF_CNT);
  unsigned* lidx   = (unsigned*)(ws + OFF_LIDX);
  float*    lval   = (float*)(ws + OFF_LVAL);
  unsigned* ccol   = (unsigned*)(ws + OFF_CCOL);
  float*    cval   = (float*)(ws + OFF_CVAL);

  prep_xc<<<2048, 256, 0, stream>>>(x, bdec, xc);
  conv_bf16<<<4096, 256, 0, stream>>>(W0, w0b, (long)S0N * ADIM);
  conv_bf16<<<8192, 256, 0, stream>>>(W1, w1b, (long)DICT * ADIM);
  transpose_wd<<<dim3(DICT / 32, ADIM / 32), dim3(32, 8), 0, stream>>>(Wd, wdtb);
  init_zero<<<16, 256, 0, stream>>>(hist, rowCnt, rowFl, cnt);

  sae_encoder_gemm<false><<<dim3(S0N / BN, BATCH / BM), GTHREADS, 0, stream>>>(
      xc, w0b, b0, nullptr, l0buf, nullptr, nullptr, S0N);
  sae_encoder_gemm<true><<<dim3(DICT / BN, BATCH / BM), GTHREADS, 0, stream>>>(
      xc, w1b, b1, l0buf, nullptr, finalB, hist, DICT);

  scan_hist<<<1, 32, 0, stream>>>(hist, cnt);
  select_topk<<<8192, 256, 0, stream>>>((const uint4*)finalB, cnt, rowCnt, lidx, lval);
  scan_rows<<<1, 32, 0, stream>>>(rowCnt, rowSt);
  scatter_csr<<<1024, 256, 0, stream>>>(cnt, lidx, lval, rowSt, rowFl, ccol, cval);
  decode_rows<<<BATCH, 256, 0, stream>>>(rowSt, rowCnt, ccol, cval, wdtb, bdec, out);
}
